// SimpleGCN_Layer_80264348828247
// MI455X (gfx1250) — matrix operand order repaired and checked by reading, not compiled, not run on hardware
//
#include <hip/hip_runtime.h>
#include <hip/hip_bf16.h>

// ---------------- types ----------------
typedef __bf16 bf16_t;
typedef __bf16 v16bf __attribute__((ext_vector_type(16)));
typedef __bf16 v8bf  __attribute__((ext_vector_type(8)));
typedef float  v8f   __attribute__((ext_vector_type(8)));

// ---------------- problem constants (match reference) ----------------
constexpr int   N_    = 50000;
constexpr int   E_    = 800000;
constexpr int   F_    = 256;
constexpr int   B_    = 100;
constexpr int   MAXN_ = 512;
constexpr int   GSZ_  = 500;          // N_/B_
constexpr int   OUTD_ = 3 * F_;       // 768
constexpr float EPS_  = 1e-5f;

// ======================================================================
// 0. zero the dense output [B, MAXN, 3F]
// ======================================================================
__global__ void k_zero_out(float* __restrict__ out, int n4) {
  int t = blockIdx.x * blockDim.x + threadIdx.x;
  if (t < n4) {
    float4 z = {0.f, 0.f, 0.f, 0.f};
    reinterpret_cast<float4*>(out)[t] = z;
  }
}

// ======================================================================
// 1. degree / dinv
// ======================================================================
__global__ void k_init_deg(float* __restrict__ deg) {
  int i = blockIdx.x * blockDim.x + threadIdx.x;
  if (i < N_) deg[i] = 1.0f;   // self loop
}

__global__ void k_deg_atomic(const int* __restrict__ dst, float* __restrict__ deg) {
  int e = blockIdx.x * blockDim.x + threadIdx.x;
  if (e < E_) atomicAdd(&deg[dst[e]], 1.0f);
}

__global__ void k_dinv(float* __restrict__ deg) {
  int i = blockIdx.x * blockDim.x + threadIdx.x;
  if (i < N_) {
    float d = deg[i];
    deg[i] = d > 0.f ? rsqrtf(d) : 0.f;
  }
}

// ======================================================================
// 2. f32 -> bf16 hi/lo split (activation matrix, row-major, no transpose)
// ======================================================================
__global__ void k_split_x(const float* __restrict__ X,
                          bf16_t* __restrict__ Hhi, bf16_t* __restrict__ Hlo,
                          int total) {
  int t = blockIdx.x * blockDim.x + threadIdx.x;
  if (t < total) {
    float v = X[t];
    bf16_t hi = (bf16_t)v;
    Hhi[t] = hi;
    Hlo[t] = (bf16_t)(v - (float)hi);
  }
}

// W [F,F] row-major -> transposed bf16 hi/lo [n][k]
__global__ void k_split_w_t(const float* __restrict__ W,
                            bf16_t* __restrict__ WThi, bf16_t* __restrict__ WTlo) {
  int t = blockIdx.x * blockDim.x + threadIdx.x;   // 256*256
  if (t < F_ * F_) {
    int k = t >> 8, n = t & 255;
    float v = W[t];                                 // W[k][n], coalesced read
    bf16_t hi = (bf16_t)v;
    WThi[n * F_ + k] = hi;
    WTlo[n * F_ + k] = (bf16_t)(v - (float)hi);
  }
}

// ======================================================================
// 3. GEMM  Y[N,256] = H[N,256] @ W[256,256]   (bf16 split, f32 accumulate)
//    per wave: 16(M) x 64(N) tile, K-loop of 8 x (K=32)
// ======================================================================
__global__ __launch_bounds__(256) void k_gemm_wmma(
    const bf16_t* __restrict__ Ahi, const bf16_t* __restrict__ Alo,
    const bf16_t* __restrict__ BThi, const bf16_t* __restrict__ BTlo,
    float* __restrict__ Y, int nrows) {
  int wave = blockIdx.x * (blockDim.x >> 5) + (threadIdx.x >> 5);
  int lane = threadIdx.x & 31;
  int mTiles = nrows >> 4;
  if (wave >= mTiles * 4) return;                 // wave-uniform exit

  int mBase = (wave >> 2) << 4;
  int nBase = (wave & 3) << 6;
  int lo16  = lane & 15;
  int half  = lane >> 4;

  const bf16_t* aHiRow = Ahi + (size_t)(mBase + lo16) * F_;
  const bf16_t* aLoRow = Alo + (size_t)(mBase + lo16) * F_;

  v8f acc[4] = {};

  union V16 { v16bf v; v8bf h[2]; };

  for (int kt = 0; kt < 8; ++kt) {
    // A 16x32 k order used here (contiguous, same as B): lanes<16 hold K 0..15, lanes>=16 hold K 16..31
    int ka = kt * 32 + half * 16;
    V16 aH, aL;
    aH.h[0] = *(const v8bf*)(aHiRow + ka);
    aH.h[1] = *(const v8bf*)(aHiRow + ka + 8);
    aL.h[0] = *(const v8bf*)(aLoRow + ka);
    aL.h[1] = *(const v8bf*)(aLoRow + ka + 8);

    // B 32x16 k order used here (contiguous, same as A): lanes<16 hold K 0..15, lanes>=16 hold K 16..31 (col = lane&15)
    int kb = kt * 32 + half * 16;
#pragma unroll
    for (int nt = 0; nt < 4; ++nt) {
      const bf16_t* bHiRow = BThi + (size_t)(nBase + nt * 16 + lo16) * F_;
      const bf16_t* bLoRow = BTlo + (size_t)(nBase + nt * 16 + lo16) * F_;
      v16bf bH = *(const v16bf*)(bHiRow + kb);
      v16bf bL = *(const v16bf*)(bLoRow + kb);
      acc[nt] = __builtin_amdgcn_wmma_f32_16x16x32_bf16(
          false, aH.v, false, bH, (short)0, acc[nt], false, false);
      acc[nt] = __builtin_amdgcn_wmma_f32_16x16x32_bf16(
          false, aH.v, false, bL, (short)0, acc[nt], false, false);
      acc[nt] = __builtin_amdgcn_wmma_f32_16x16x32_bf16(
          false, aL.v, false, bH, (short)0, acc[nt], false, false);
    }
  }

  // C/D f32 layout: VGPR r, lane l -> row = r + 8*(l>>4), col = l&15
  int rowOut0 = mBase + half * 8;
#pragma unroll
  for (int nt = 0; nt < 4; ++nt) {
    int col = nBase + nt * 16 + lo16;
#pragma unroll
    for (int r = 0; r < 8; ++r)
      Y[(size_t)(rowOut0 + r) * F_ + col] = acc[nt][r];
  }
}

// ======================================================================
// 4. S = bias + self-loop contribution  (norm_ii = dinv[i]^2)
// ======================================================================
__global__ void k_self_bias(const float* __restrict__ Y, const float* __restrict__ dinv,
                            const float* __restrict__ bias, float* __restrict__ S) {
  int t = blockIdx.x * blockDim.x + threadIdx.x;   // N_*64
  if (t >= N_ * 64) return;
  int i = t >> 6;
  int f = (t & 63) << 2;
  float w = dinv[i] * dinv[i];
  const float4 y = reinterpret_cast<const float4*>(Y)[(size_t)i * 64 + (t & 63)];
  float4 b = reinterpret_cast<const float4*>(bias)[t & 63];
  float4 s;
  s.x = b.x + y.x * w; s.y = b.y + y.y * w;
  s.z = b.z + y.z * w; s.w = b.w + y.w * w;
  reinterpret_cast<float4*>(S)[(size_t)i * 64 + (t & 63)] = s;
  (void)f;
}

// ======================================================================
// 5. edge scatter: S[dst] += Y[src] * dinv[src]*dinv[dst]
// ======================================================================
__global__ void k_edge_scatter(const int* __restrict__ src, const int* __restrict__ dst,
                               const float* __restrict__ dinv,
                               const float* __restrict__ Y, float* __restrict__ S) {
  long long t = (long long)blockIdx.x * blockDim.x + threadIdx.x;  // E_*64
  if (t >= (long long)E_ * 64) return;
  int e = (int)(t >> 6);
  int q = (int)(t & 63);
  int s = src[e], d = dst[e];
  float nrm = dinv[s] * dinv[d];
  float4 y = reinterpret_cast<const float4*>(Y)[(size_t)s * 64 + q];
  float* o = S + (size_t)d * F_ + (q << 2);
  atomicAdd(o + 0, y.x * nrm);
  atomicAdd(o + 1, y.y * nrm);
  atomicAdd(o + 2, y.z * nrm);
  atomicAdd(o + 3, y.w * nrm);
}

// ======================================================================
// 6. BN stats
// ======================================================================
__global__ void k_zero_sums(float* __restrict__ sums) {
  int t = blockIdx.x * blockDim.x + threadIdx.x;
  if (t < 2 * F_) sums[t] = 0.f;
}

__global__ __launch_bounds__(256) void k_bn_stats(const float* __restrict__ S,
                                                  float* __restrict__ sum,
                                                  float* __restrict__ sumsq) {
  int f = threadIdx.x;                 // 0..255 (== F_)
  int r0 = blockIdx.x * 250;           // 200 blocks x 250 rows = 50000
  float s = 0.f, q = 0.f;
  for (int r = 0; r < 250; ++r) {
    float v = S[(size_t)(r0 + r) * F_ + f];
    s += v;
    q += v * v;
  }
  atomicAdd(&sum[f], s);
  atomicAdd(&sumsq[f], q);
}

// ======================================================================
// 7. BN apply + ReLU + bf16 re-split for next layer + dense output write
// ======================================================================
__global__ void k_bn_apply(const float* __restrict__ S,
                           const float* __restrict__ sum, const float* __restrict__ sumsq,
                           const float* __restrict__ gamma, const float* __restrict__ beta,
                           bf16_t* __restrict__ Hhi, bf16_t* __restrict__ Hlo,
                           float* __restrict__ out, int layer) {
  int t = blockIdx.x * blockDim.x + threadIdx.x;   // N_*64
  if (t >= N_ * 64) return;
  int i = t >> 6;
  int f0 = (t & 63) << 2;
  int g = i / GSZ_;
  int p = i - g * GSZ_;
  const float invN = 1.0f / (float)N_;
  float* op = out + (size_t)g * MAXN_ * OUTD_ + (size_t)p * OUTD_ + layer * F_ + f0;
#pragma unroll
  for (int j = 0; j < 4; ++j) {
    int f = f0 + j;
    float mu  = sum[f] * invN;
    float var = sumsq[f] * invN - mu * mu;
    float inv = rsqrtf(var + EPS_);
    float h = (S[(size_t)i * F_ + f] - mu) * inv * gamma[f] + beta[f];
    h = fmaxf(h, 0.f);
    bf16_t hi = (bf16_t)h;
    Hhi[(size_t)i * F_ + f] = hi;
    Hlo[(size_t)i * F_ + f] = (bf16_t)(h - (float)hi);
    op[j] = h;
  }
}

// ======================================================================
// launch
// ======================================================================
extern "C" void kernel_launch(void* const* d_in, const int* in_sizes, int n_in,
                              void* d_out, int out_size, void* d_ws, size_t ws_size,
                              hipStream_t stream) {
  const float* x        = (const float*)d_in[0];
  const int*   eidx     = (const int*)d_in[1];     // [2,E] flat: src then dst
  const int*   e_src    = eidx;
  const int*   e_dst    = eidx + E_;

  char* ws = (char*)d_ws;
  size_t off = 0;
  auto alloc = [&](size_t bytes) -> char* {
    char* p = ws + off;
    off += (bytes + 255) & ~(size_t)255;
    return p;
  };
  float*  dinv  = (float*)alloc((size_t)N_ * 4);
  bf16_t* Hhi   = (bf16_t*)alloc((size_t)N_ * F_ * 2);
  bf16_t* Hlo   = (bf16_t*)alloc((size_t)N_ * F_ * 2);
  bf16_t* WThi  = (bf16_t*)alloc((size_t)F_ * F_ * 2);
  bf16_t* WTlo  = (bf16_t*)alloc((size_t)F_ * F_ * 2);
  float*  Y     = (float*)alloc((size_t)N_ * F_ * 4);
  float*  S     = (float*)alloc((size_t)N_ * F_ * 4);
  float*  sums  = (float*)alloc((size_t)2 * F_ * 4);   // [sum | sumsq]
  float*  sum   = sums;
  float*  sumsq = sums + F_;
  (void)ws_size; (void)in_sizes; (void)n_in; (void)out_size;

  const int T = 256;

  // dense output zero:  B*MAXN*3F floats
  {
    int n4 = (B_ * MAXN_ * OUTD_) / 4;
    k_zero_out<<<(n4 + T - 1) / T, T, 0, stream>>>((float*)d_out, n4);
  }

  // degree -> dinv
  k_init_deg  <<<(N_ + T - 1) / T, T, 0, stream>>>(dinv);
  k_deg_atomic<<<(E_ + T - 1) / T, T, 0, stream>>>(e_dst, dinv);
  k_dinv      <<<(N_ + T - 1) / T, T, 0, stream>>>(dinv);

  // x -> bf16 hi/lo
  k_split_x<<<((N_ * F_) + T - 1) / T, T, 0, stream>>>(x, Hhi, Hlo, N_ * F_);

  const int gemmWaves  = (N_ / 16) * 4;            // 12500
  const int gemmBlocks = (gemmWaves + 7) / 8;      // 8 waves per 256-thread block

  for (int l = 0; l < 3; ++l) {
    const float* W     = (const float*)d_in[4 + 4 * l + 0];
    const float* bias  = (const float*)d_in[4 + 4 * l + 1];
    const float* gamma = (const float*)d_in[4 + 4 * l + 2];
    const float* beta  = (const float*)d_in[4 + 4 * l + 3];

    k_split_w_t<<<(F_ * F_ + T - 1) / T, T, 0, stream>>>(W, WThi, WTlo);

    k_gemm_wmma<<<gemmBlocks, T, 0, stream>>>(Hhi, Hlo, WThi, WTlo, Y, N_);

    k_self_bias<<<((N_ * 64) + T - 1) / T, T, 0, stream>>>(Y, dinv, bias, S);

    {
      long long tot = (long long)E_ * 64;
      int blocks = (int)((tot + T - 1) / T);
      k_edge_scatter<<<blocks, T, 0, stream>>>(e_src, e_dst, dinv, Y, S);
    }

    k_zero_sums<<<(2 * F_ + T - 1) / T, T, 0, stream>>>(sums);
    k_bn_stats <<<200, T, 0, stream>>>(S, sum, sumsq);

    k_bn_apply<<<((N_ * 64) + T - 1) / T, T, 0, stream>>>(
        S, sum, sumsq, gamma, beta, Hhi, Hlo, (float*)d_out, l);
  }
}
